// RbfNet_19842748908183
// MI455X (gfx1250) — compile-verified, matrix operand order checked by reading, not run on hardware
//
#include <hip/hip_runtime.h>

typedef __attribute__((ext_vector_type(16))) __bf16 v16bf;
typedef __attribute__((ext_vector_type(8)))  float  v8f;

union ABu { uint4 q[2]; v16bf v; };

__device__ __forceinline__ unsigned short f2bf(float f) {
    unsigned int u = __float_as_uint(f);
    unsigned int r = u + 0x7FFFu + ((u >> 16) & 1u);   // round-to-nearest-even
    return (unsigned short)(r >> 16);
}
__device__ __forceinline__ float bf2f(unsigned short h) {
    return __uint_as_float(((unsigned int)h) << 16);
}

// ---------------------------------------------------------------------------
// Pack activations -> bf16 [Npad][64], zero-padded in both rows and channels.
// ---------------------------------------------------------------------------
__global__ void pack_x_kernel(const float* __restrict__ a,
                              unsigned short* __restrict__ xb,
                              int N, int Npad, int Cin, int stride_a, int do_relu) {
    long t = (long)blockIdx.x * blockDim.x + threadIdx.x;
    if (t >= (long)Npad * 64) return;
    int i = (int)(t >> 6), c = (int)(t & 63);
    float val = 0.f;
    if (i < N && c < Cin) {
        val = a[(long)i * stride_a + c];
        if (do_relu) val = fmaxf(val, 0.f);
    }
    xb[t] = f2bf(val);
}

// ---------------------------------------------------------------------------
// Pack conv weights cw[K=64][Cin][Cout] -> Wp[m][cin(64, zero-pad)] with
// m = k*COchunk + co  (k-major so z rows are [k][co] per node: 4-tap friendly)
// ---------------------------------------------------------------------------
__global__ void pack_w_kernel(const float* __restrict__ cw,
                              unsigned short* __restrict__ Wp,
                              int Cin, int Cout_total, int co0, int COchunk) {
    long total = (long)COchunk * 64 * 64;
    long t = (long)blockIdx.x * blockDim.x + threadIdx.x;
    if (t >= total) return;
    int m   = (int)(t >> 6);
    int cin = (int)(t & 63);
    int k   = m / COchunk;
    int co  = m - k * COchunk;
    float val = (cin < Cin)
        ? cw[((long)k * Cin + cin) * Cout_total + (co0 + co)] : 0.f;
    Wp[t] = f2bf(val);
}

// ---------------------------------------------------------------------------
// z[n][m] = xb[n][:] @ Wp[m][:]   via v_wmma_f32_16x16x32_bf16
// grid.x = Npad/16 node tiles, grid.y = Mtotal/128, 8 waves/block (one 16-col
// tile each). K-loop over ksteps chunks of 32 bf16.
// ---------------------------------------------------------------------------
__global__ __launch_bounds__(256) void gemm_z_kernel(
    const unsigned short* __restrict__ xb,   // [Npad][64]
    const unsigned short* __restrict__ Wp,   // [Mtotal][64]
    unsigned short* __restrict__ z,          // [Npad][Mtotal]
    int Mtotal, int ksteps) {
    const int wave = threadIdx.x >> 5;
    const int lane = threadIdx.x & 31;
    const int  mtile = (blockIdx.y * 8 + wave) * 16;
    const long ntile = (long)blockIdx.x * 16;
    const int row = lane & 15;           // A: M row / B: output column
    const int kh  = (lane >> 4) * 8;     // per-lane K group offset

    const unsigned short* arow = xb + (ntile + row) * 64 + kh;
    const unsigned short* brow = Wp + ((long)(mtile + row)) * 64 + kh;

    v8f acc = {};
    for (int s = 0; s < ksteps; ++s) {
        ABu a, b;
        a.q[0] = *(const uint4*)(arow + s * 32);
        a.q[1] = *(const uint4*)(arow + s * 32 + 16);
        b.q[0] = *(const uint4*)(brow + s * 32);
        b.q[1] = *(const uint4*)(brow + s * 32 + 16);
        acc = __builtin_amdgcn_wmma_f32_16x16x32_bf16(
            false, a.v, false, b.v, (short)0, acc, false, false);
    }
    // C/D layout: lane l, vgpr r -> (M = r + 8*(l>=16), N = l&15)
    const int  ncol = mtile + row;
    const long nodebase = ntile + (lane >> 4) * 8;
#pragma unroll
    for (int r = 0; r < 8; ++r)
        z[(nodebase + r) * (long)Mtotal + ncol] = f2bf(acc[r]);
}

// ---------------------------------------------------------------------------
// base: out[i, col0+co] = fb[co] (+cb[co]) (+resid) + sum_cin act(a)*fw
// Fully (over)writes its region -> atomics in edge kernel land on fresh data.
// ---------------------------------------------------------------------------
__global__ void base_kernel(const float* __restrict__ a, int stride_a, int do_relu,
                            const float* __restrict__ fw, const float* __restrict__ fb,
                            const float* __restrict__ cb, const float* __restrict__ resid,
                            float* __restrict__ out,
                            int N, int Cin, int Cout, int col0, int out_stride) {
    long t = (long)blockIdx.x * blockDim.x + threadIdx.x;
    if (t >= (long)N * Cout) return;
    long i  = t / Cout;
    int  co = (int)(t - i * Cout);
    float s = fb[co];
    if (cb)    s += cb[co];
    if (resid) s += resid[i * out_stride + col0 + co];
    for (int cin = 0; cin < Cin; ++cin) {
        float av = a[i * stride_a + cin];
        if (do_relu) av = fmaxf(av, 0.f);
        s = fmaf(av, fw[(long)cin * Cout + co], s);
    }
    out[i * out_stride + col0 + co] = s;
}

// ---------------------------------------------------------------------------
// Edge scatter, 4-tap form. The 8x8 separable hat basis is bilinear
// interpolation on a uniform grid: only (iu,iv),(iu,iv+1),(iu+1,iv),
// (iu+1,iv+1) are nonzero. Per lane: 4 bf16 gathers, 4 FMAs, 1 f32 atomic.
// z layout [node][k=ku*8+kv][co] -> taps are coalesced across co lanes.
// ---------------------------------------------------------------------------
__global__ __launch_bounds__(256) void edge_conv_kernel(
    const int* __restrict__ ei, const int* __restrict__ ej,
    const float* __restrict__ pos,
    const unsigned short* __restrict__ z,    // [Npad][64*CO]
    float* __restrict__ out,
    int CO, int out_stride, int out_col0, int E) {
    const int lane = threadIdx.x & 31;
    const int wave = threadIdx.x >> 5;
    const int epw  = 32 / CO;                // edges per wave
    const int sub  = lane / CO;
    const int co   = lane - sub * CO;
    const long e = (long)blockIdx.x * (8 * epw) + wave * epw + sub;
    if (e >= E) return;
    const int i = ei[e], j = ej[e];
    if (i == j) return;                      // centerIgnore mask

    float dx = pos[2 * i]     - pos[2 * j];  // SUPPORT = 1.0
    float dy = pos[2 * i + 1] - pos[2 * j + 1];
    dx = fminf(1.f, fmaxf(-1.f, dx));
    dy = fminf(1.f, fmaxf(-1.f, dy));
    float r = sqrtf(dx * dx + dy * dy + 1e-12f);
    float u = fminf(1.f, fmaxf(-1.f, 2.f * r - 1.f));
    float v = atan2f(dy, dx) * 0.3183098861837907f;   // 1/pi

    // bilinear cell + fractions on the 8-center grid (spacing 2/7)
    float up = (u + 1.f) * 3.5f;             // in [0,7]
    float vp = (v + 1.f) * 3.5f;
    int iu = min(6, (int)up);
    int iv = min(6, (int)vp);
    float fu = up - (float)iu;
    float fv = vp - (float)iv;
    float w00 = (1.f - fu) * (1.f - fv);
    float w01 = (1.f - fu) * fv;
    float w10 = fu * (1.f - fv);
    float w11 = fu * fv;

    const unsigned short* zr =
        z + (long)j * (64 * CO) + (iu * 8 + iv) * CO + co;
    float sum = w00 * bf2f(zr[0])
              + w01 * bf2f(zr[CO])           // k+1  (iv+1)
              + w10 * bf2f(zr[8 * CO])       // k+8  (iu+1)
              + w11 * bf2f(zr[9 * CO]);      // k+9

    atomicAdd(out + (long)i * out_stride + out_col0 + co, sum);
}

__global__ void scale_kernel(float* __restrict__ out, long n, float s) {
    long t = (long)blockIdx.x * blockDim.x + threadIdx.x;
    if (t < n) out[t] *= s;
}

// ---------------------------------------------------------------------------
extern "C" void kernel_launch(void* const* d_in, const int* in_sizes, int n_in,
                              void* d_out, int out_size, void* d_ws, size_t ws_size,
                              hipStream_t stream) {
    const float* pos  = (const float*)d_in[0];
    const float* feat = (const float*)d_in[1];
    const int*   ei   = (const int*)d_in[2];
    const int*   ej   = (const int*)d_in[3];
    const float *cw[4], *cb[4], *fw[4], *fb[4];
    for (int l = 0; l < 4; ++l) {
        cw[l] = (const float*)d_in[4 + 4 * l + 0];
        cb[l] = (const float*)d_in[4 + 4 * l + 1];
        fw[l] = (const float*)d_in[4 + 4 * l + 2];
        fb[l] = (const float*)d_in[4 + 4 * l + 3];
    }
    const int N    = in_sizes[0] / 2;
    const int E    = in_sizes[2];
    const int Npad = (N + 15) & ~15;

    char* ws = (char*)d_ws;
    size_t off = 0;
    float* ansA = (float*)(ws + off); off += (size_t)N * 64 * 4;
    float* ansB = (float*)(ws + off); off += (size_t)N * 64 * 4;
    unsigned short* xb = (unsigned short*)(ws + off); off += (size_t)Npad * 64 * 2;
    unsigned short* Wp = (unsigned short*)(ws + off); off += (size_t)2048 * 64 * 2;
    unsigned short* z  = (unsigned short*)(ws + off); off += (size_t)Npad * 2048 * 2;
    float* out = (float*)d_out;

    auto cdiv = [](long a, long b) { return (int)((a + b - 1) / b); };

    auto run_chunk = [&](const float* cwl, int Cin, int Cout_total, int co0,
                         int COchunk, int ksteps, float* dst, int dst_stride,
                         int dst_col0) {
        const int Mtotal = COchunk * 64;
        pack_w_kernel<<<cdiv((long)Mtotal * 64, 256), 256, 0, stream>>>(
            cwl, Wp, Cin, Cout_total, co0, COchunk);
        dim3 g(Npad / 16, Mtotal / 128 ? Mtotal / 128 : 1);
        gemm_z_kernel<<<g, 256, 0, stream>>>(xb, Wp, z, Mtotal, ksteps);
        const int per = 8 * (32 / COchunk);
        edge_conv_kernel<<<cdiv(E, per), 256, 0, stream>>>(
            ei, ej, pos, z, dst, COchunk, dst_stride, dst_col0, E);
    };

    // ---- Layer 0: Cin=4 (raw features), Cout=32; ans = [lin0 | conv0] ----
    pack_x_kernel<<<cdiv((long)Npad * 64, 256), 256, 0, stream>>>(
        feat, xb, N, Npad, 4, 4, 0);
    base_kernel<<<cdiv((long)N * 32, 256), 256, 0, stream>>>(
        feat, 4, 0, fw[0], fb[0], nullptr, nullptr, ansA, N, 4, 32, 0, 64);
    base_kernel<<<cdiv((long)N * 32, 256), 256, 0, stream>>>(   // cols 32..63 = cb0
        feat, 4, 0, fw[0], cb[0], nullptr, nullptr, ansA, N, 0, 32, 32, 64);
    run_chunk(cw[0], 4, 32, 0, 32, /*ksteps=*/1, ansA, 64, 32);

    // ---- Layer 1: relu(ansA) -> ansB, Cin=64, Cout=64, no residual ----
    pack_x_kernel<<<cdiv((long)Npad * 64, 256), 256, 0, stream>>>(
        ansA, xb, N, Npad, 64, 64, 1);
    base_kernel<<<cdiv((long)N * 64, 256), 256, 0, stream>>>(
        ansA, 64, 1, fw[1], fb[1], cb[1], nullptr, ansB, N, 64, 64, 0, 64);
    run_chunk(cw[1], 64, 64, 0,  32, 2, ansB, 64, 0);
    run_chunk(cw[1], 64, 64, 32, 32, 2, ansB, 64, 32);

    // ---- Layer 2: relu(ansB) -> ansA, residual += ansB ----
    pack_x_kernel<<<cdiv((long)Npad * 64, 256), 256, 0, stream>>>(
        ansB, xb, N, Npad, 64, 64, 1);
    base_kernel<<<cdiv((long)N * 64, 256), 256, 0, stream>>>(
        ansB, 64, 1, fw[2], fb[2], cb[2], ansB, ansA, N, 64, 64, 0, 64);
    run_chunk(cw[2], 64, 64, 0,  32, 2, ansA, 64, 0);
    run_chunk(cw[2], 64, 64, 32, 32, 2, ansA, 64, 32);

    // ---- Layer 3: relu(ansA) -> d_out, Cin=64, Cout=2 ----
    pack_x_kernel<<<cdiv((long)Npad * 64, 256), 256, 0, stream>>>(
        ansA, xb, N, Npad, 64, 64, 1);
    base_kernel<<<cdiv((long)N * 2, 256), 256, 0, stream>>>(
        ansA, 64, 1, fw[3], fb[3], cb[3], nullptr, out, N, 64, 2, 0, 2);
    run_chunk(cw[3], 64, 2, 0, 2, 2, out, 2, 0);

    scale_kernel<<<cdiv((long)N * 2, 256), 256, 0, stream>>>(
        out, (long)N * 2, 1.f / 128.f);
}
